// SentencePairSLSTMEncoderZY_75101798138243
// MI455X (gfx1250) — compile-verified, matrix operand order checked by reading, not run on hardware
//
#include <hip/hip_runtime.h>
#include <hip/hip_bf16.h>

#define H 1024
#define BB 32
#define SS 128          // S1 == S2 == 128
#define RSEQ (BB*SS)    // 4096 rows per sequence
#define K4 (4*H)        // 4096
#define N6 (6*H)        // 6144
#define EPSV 0.001f
#define NEGV 1e25f

// ---------------------------------------------------------------- bf16 utils
typedef __attribute__((ext_vector_type(16))) __bf16 v16bf;
typedef __attribute__((ext_vector_type(8)))  float  v8f;

union Frag { uint4 q[2]; v16bf v; };

__device__ __forceinline__ unsigned short f2bf(float f) {
  unsigned int u = __float_as_uint(f);
  unsigned int r = (u + 0x7FFFu + ((u >> 16) & 1u)) >> 16;
  return (unsigned short)r;
}

// ---------------------------------------------------------------- weight packing
struct P18 { const float* p[18]; };   // per-gate Wx, Wh, Wi
struct P6  { const float* p[6];  };

// WbigT[n*K4 + k] = Wstack_g[k, n%H], g = n/H, K blocks: [Wx | Wh(0:H) | Wh(H:2H) | Wi]
__global__ void pack_wbigT(unsigned short* __restrict__ dst, P18 w) {
  int idx = blockIdx.x * blockDim.x + threadIdx.x;
  if (idx >= N6 * K4) return;
  int n = idx / K4, k = idx % K4;
  int g = n >> 10, c = n & 1023;
  int kb = k >> 10, kc = k & 1023;
  const float* Wx = w.p[g*3+0]; const float* Wh = w.p[g*3+1]; const float* Wi = w.p[g*3+2];
  float v;
  if      (kb == 0) v = Wx[kc*H + c];
  else if (kb == 1) v = Wh[kc*H + c];
  else if (kb == 2) v = Wh[(kc+H)*H + c];
  else              v = Wi[kc*H + c];
  dst[idx] = f2bf(v);
}

// dst[n*K + k] = p[n/H][k*H + n%H]   (transpose+convert of column-stacked weights)
__global__ void pack_colT(unsigned short* __restrict__ dst, P6 w, int N, int K) {
  int idx = blockIdx.x * blockDim.x + threadIdx.x;
  if (idx >= N * K) return;
  int n = idx / K, k = idx % K;
  dst[idx] = f2bf(w.p[n >> 10][(size_t)k * H + (n & 1023)]);
}

// ---------------------------------------------------------------- WMMA GEMM
// C[M,N] (f32, ldc) = A[M,K] (bf16, lda) * WT[N,K] (bf16)   [+ bias2d[(m/rps)*biasld + n]] [+=]
// 8 waves / block, 2 (M) x 4 (N); wave tile 64x64 = 4x4 WMMA accumulators.
// Block tile 128 x 256.  ~32 FLOP per L2 byte at register level.
__global__ __launch_bounds__(256)
void gemm_bf16_wmma(const unsigned short* __restrict__ A, int lda,
                    const unsigned short* __restrict__ WT,
                    float* __restrict__ C, int ldc,
                    int M, int N, int K,
                    const float* __restrict__ bias2d, int biasld, int rps,
                    int accumulate)
{
  const int lane = threadIdx.x & 31;
  const int wave = threadIdx.x >> 5;
  const int wm = wave & 1;       // 2 waves in M
  const int wn = wave >> 1;      // 4 waves in N
  const int row0 = blockIdx.y * 128 + wm * 64;
  const int col0 = blockIdx.x * 256 + wn * 64;
  const int lr   = lane & 15;
  const int half = lane >> 4;

  v8f acc[4][4] = {};

  for (int k = 0; k < K; k += 32) {
    Frag a[4];
#pragma unroll
    for (int mt = 0; mt < 4; ++mt) {
      int m = row0 + mt*16 + lr; if (m >= M) m = M - 1;
      const unsigned short* p = A + (size_t)m * lda + k + half*8;
      a[mt].q[0] = *(const uint4*)(p);
      a[mt].q[1] = *(const uint4*)(p + 16);
    }
    Frag b[4];
#pragma unroll
    for (int nt = 0; nt < 4; ++nt) {
      int n = col0 + nt*16 + lr; if (n >= N) n = N - 1;
      const unsigned short* p = WT + (size_t)n * K + k + half*8;
      b[nt].q[0] = *(const uint4*)(p);
      b[nt].q[1] = *(const uint4*)(p + 16);
    }
#pragma unroll
    for (int mt = 0; mt < 4; ++mt)
#pragma unroll
      for (int nt = 0; nt < 4; ++nt)
        acc[mt][nt] = __builtin_amdgcn_wmma_f32_16x16x32_bf16(
            false, a[mt].v, false, b[nt].v, (short)0, acc[mt][nt], false, false);
  }

#pragma unroll
  for (int mt = 0; mt < 4; ++mt) {
#pragma unroll
    for (int nt = 0; nt < 4; ++nt) {
      int n = col0 + nt*16 + lr;
#pragma unroll
      for (int v = 0; v < 8; ++v) {
        int m = row0 + mt*16 + v + half*8;
        if (m < M && n < N) {
          float val = acc[mt][nt][v];
          if (bias2d) val += bias2d[(size_t)(m / rps) * biasld + n];
          size_t o = (size_t)m * ldc + n;
          C[o] = accumulate ? (C[o] + val) : val;
        }
      }
    }
  }
}

// ---------------------------------------------------------------- LayerNorm + sigmoid
// out[row*outld+outoff+c] = sigmoid( LN(in[row*ld+coloff+ :H]) * gamma + beta + bias )
__global__ __launch_bounds__(256)
void ln_sigmoid(const float* __restrict__ in, int ld, int coloff,
                float* __restrict__ out, int outld, int outoff,
                const float* __restrict__ gamma, const float* __restrict__ beta,
                const float* __restrict__ bias)
{
  __shared__ float red[256];
  const int row = blockIdx.x;
  const int t = threadIdx.x;
  const float* src = in + (size_t)row * ld + coloff;
  float v[4];
  float s = 0.f;
#pragma unroll
  for (int i = 0; i < 4; ++i) { v[i] = src[t + i*256]; s += v[i]; }
  red[t] = s; __syncthreads();
  for (int w = 128; w > 0; w >>= 1) { if (t < w) red[t] += red[t + w]; __syncthreads(); }
  float mu = red[0] * (1.f / H); __syncthreads();
  s = 0.f;
#pragma unroll
  for (int i = 0; i < 4; ++i) { float d = v[i] - mu; s += d * d; }
  red[t] = s; __syncthreads();
  for (int w = 128; w > 0; w >>= 1) { if (t < w) red[t] += red[t + w]; __syncthreads(); }
  float rstd = rsqrtf(red[0] * (1.f / H) + EPSV);
  float* dst = out + (size_t)row * outld + outoff;
#pragma unroll
  for (int i = 0; i < 4; ++i) {
    int c = t + i*256;
    float x = (v[i] - mu) * rstd * gamma[c] + beta[c] + bias[c];
    dst[c] = 1.f / (1.f + expf(-x));
  }
}

// ---------------------------------------------------------------- state init
__global__ void init_state(const float* __restrict__ inp, const float* __restrict__ h0,
                           const float* __restrict__ c0, const float* __restrict__ mask,
                           float* __restrict__ h, float* __restrict__ c,
                           float* __restrict__ emb)
{
  int idx = blockIdx.x * blockDim.x + threadIdx.x;
  if (idx >= RSEQ * H) return;
  int r = idx >> 10;
  float m = mask[r];
  h[idx] = h0[idx] * m; c[idx] = c0[idx] * m; emb[idx] = inp[idx] * m;
}

__global__ void init_global(const float* __restrict__ h1, const float* __restrict__ h2,
                            const float* __restrict__ c1, const float* __restrict__ c2,
                            float* __restrict__ gh, float* __restrict__ gc)
{
  int idx = blockIdx.x * blockDim.x + threadIdx.x;
  if (idx >= BB * H) return;
  int b = idx >> 10, c = idx & 1023;
  float sh = 0.f, sc = 0.f;
  for (int s = 0; s < SS; ++s) {
    size_t o = (size_t)(b*SS + s) * H + c;
    sh += h1[o] + h2[o]; sc += c1[o] + c2[o];
  }
  gh[idx] = sh * (1.f / (2*SS)); gc[idx] = sc * (1.f / (2*SS));
}

// ---------------------------------------------------------------- fusion (comb / feats)
__global__ void fusion_comb(const float* __restrict__ h1, const float* __restrict__ m1,
                            const float* __restrict__ h2, const float* __restrict__ m2,
                            float* __restrict__ dst)
{
  int idx = blockIdx.x * blockDim.x + threadIdx.x;
  if (idx >= BB * H) return;
  int b = idx >> 10, c = idx & 1023;
  float mean1 = 0.f, mean2 = 0.f, max1 = -1e30f, max2 = -1e30f;
  for (int s = 0; s < SS; ++s) {
    float v1 = h1[(size_t)(b*SS + s) * H + c];
    float v2 = h2[(size_t)(b*SS + s) * H + c];
    mean1 += v1; mean2 += v2;
    if (m1[b*SS + s] > 0.f) max1 = fmaxf(max1, v1);
    if (m2[b*SS + s] > 0.f) max2 = fmaxf(max2, v2);
  }
  mean1 *= (1.f / SS); mean2 *= (1.f / SS);
  float* d = dst + (size_t)b * N6;
  d[c]        = max1;
  d[c + H]    = mean1;
  d[c + 2*H]  = max2;
  d[c + 3*H]  = mean2;
  d[c + 4*H]  = fabsf(max1 - max2);
  d[c + 5*H]  = max1 * max2;
}

// ---------------------------------------------------------------- bf16 staging
// X[r, :] = [h | hb | ha | emb] as bf16, 4096 cols
__global__ void build_X(const float* __restrict__ h, const float* __restrict__ emb,
                        unsigned short* __restrict__ X)
{
  int idx = blockIdx.x * blockDim.x + threadIdx.x;
  if (idx >= RSEQ * K4) return;
  int r = idx >> 12, c = idx & 4095;
  int s = r & (SS - 1);
  float v;
  if (c < H)            v = h[(size_t)r * H + c];
  else if (c < 2*H)     v = (s > 0)      ? h[(size_t)(r-1) * H + (c - H)]   : 0.f;
  else if (c < 3*H)     v = (s < SS-1)   ? h[(size_t)(r+1) * H + (c - 2*H)] : 0.f;
  else                  v = emb[(size_t)r * H + (c - 3*H)];
  X[idx] = f2bf(v);
}

// ghcomb[b, :] = [g_h | comb] as bf16, 7168 cols
__global__ void build_ghcomb(const float* __restrict__ gh, const float* __restrict__ comb,
                             unsigned short* __restrict__ dst)
{
  int idx = blockIdx.x * blockDim.x + threadIdx.x;
  if (idx >= BB * (H + N6)) return;
  int b = idx / (H + N6), c = idx % (H + N6);
  float v = (c < H) ? gh[(size_t)b * H + c] : comb[(size_t)b * N6 + (c - H)];
  dst[idx] = f2bf(v);
}

// ---------------------------------------------------------------- score softmax -> new g_h/g_c
__global__ void scores_update(const float* __restrict__ gf1, const float* __restrict__ gf2,
                              const float* __restrict__ gd, const float* __restrict__ go,
                              const float* __restrict__ m1, const float* __restrict__ m2,
                              const float* __restrict__ c1, const float* __restrict__ c2,
                              const float* __restrict__ gc,
                              float* __restrict__ gh_new, float* __restrict__ gc_new)
{
  int idx = blockIdx.x * blockDim.x + threadIdx.x;
  if (idx >= BB * H) return;
  int b = idx >> 10, c = idx & 1023;
  float vd = gd[idx];
  float mx = vd;
  for (int s = 0; s < SS; ++s) {
    float v = gf1[(size_t)(b*SS + s) * H + c] + (m1[b*SS + s] - 1.f) * NEGV;
    mx = fmaxf(mx, v);
    v = gf2[(size_t)(b*SS + s) * H + c] + (m2[b*SS + s] - 1.f) * NEGV;
    mx = fmaxf(mx, v);
  }
  float ed = expf(vd - mx);
  float den = ed;
  float acc = ed * gc[idx];
  for (int s = 0; s < SS; ++s) {
    size_t o = (size_t)(b*SS + s) * H + c;
    float e1 = expf(gf1[o] + (m1[b*SS + s] - 1.f) * NEGV - mx);
    float e2 = expf(gf2[o] + (m2[b*SS + s] - 1.f) * NEGV - mx);
    den += e1 + e2;
    acc += e1 * c1[o] + e2 * c2[o];
  }
  float ngc = acc / den;
  gc_new[idx] = ngc;
  gh_new[idx] = go[idx] * tanhf(ngc);
}

// ---------------------------------------------------------------- local sLSTM update
// P holds post-LN sigmoid gates [f1 f2 f3 f4 i o] at col blocks g*H (row stride 6H)
__global__ void slstm_update(const float* __restrict__ P, const float* __restrict__ emb,
                             const float* __restrict__ cold, const float* __restrict__ gc,
                             const float* __restrict__ mask,
                             float* __restrict__ hnew, float* __restrict__ cnew)
{
  int idx = blockIdx.x * blockDim.x + threadIdx.x;
  if (idx >= RSEQ * H) return;
  int r = idx >> 10, c = idx & 1023;
  int b = r >> 7, s = r & (SS - 1);
  const float* row = P + (size_t)r * N6;
  float f1 = row[c], f2 = row[c + H], f3 = row[c + 2*H], f4 = row[c + 3*H];
  float it = row[c + 4*H], ot = row[c + 5*H];
  float mx = fmaxf(fmaxf(fmaxf(f1, f2), fmaxf(f3, f4)), it);
  float e1 = expf(f1 - mx), e2 = expf(f2 - mx), e3 = expf(f3 - mx),
        e4 = expf(f4 - mx), e5 = expf(it - mx);
  float den = e1 + e2 + e3 + e4 + e5;
  float cb = (s > 0)      ? cold[(size_t)(r-1) * H + c] : 0.f;
  float ca = (s < SS - 1) ? cold[(size_t)(r+1) * H + c] : 0.f;
  float cn = (e1*cb + e2*ca + e3*emb[idx] + e4*gc[(size_t)b * H + c] + e5*cold[idx]) / den;
  float hn = ot * tanhf(cn);
  float m = mask[b*SS + s];
  hnew[idx] = hn * m;
  cnew[idx] = cn * m;
}

// ================================================================ host
static inline dim3 gemm_grid(int M, int N) { return dim3((N + 255) / 256, (M + 127) / 128); }

extern "C" void kernel_launch(void* const* d_in, const int* in_sizes, int n_in,
                              void* d_out, int out_size, void* d_ws, size_t ws_size,
                              hipStream_t stream)
{
  (void)in_sizes; (void)n_in; (void)out_size; (void)ws_size;
  const float* s1i = (const float*)d_in[0];
  const float* m1  = (const float*)d_in[1];
  const float* s2i = (const float*)d_in[2];
  const float* m2  = (const float*)d_in[3];
  const float* h10 = (const float*)d_in[4];
  const float* c10 = (const float*)d_in[5];
  const float* h20 = (const float*)d_in[6];
  const float* c20 = (const float*)d_in[7];

  const float *Wx[6], *Wh[6], *Wi[6], *Wd[6], *bg[6], *ng[6], *nb[6];
  for (int g = 0; g < 6; ++g) {
    int base = 8 + 7 * g;
    Wx[g] = (const float*)d_in[base + 0];
    Wh[g] = (const float*)d_in[base + 1];
    Wi[g] = (const float*)d_in[base + 2];
    Wd[g] = (const float*)d_in[base + 3];
    bg[g] = (const float*)d_in[base + 4];
    ng[g] = (const float*)d_in[base + 5];
    nb[g] = (const float*)d_in[base + 6];
  }
  const float* gWxd  = (const float*)d_in[50];
  const float* gWhd  = (const float*)d_in[51];
  const float* gWxo  = (const float*)d_in[52];
  const float* gWho  = (const float*)d_in[53];
  const float* gWxf1 = (const float*)d_in[54];
  const float* gWhf1 = (const float*)d_in[55];
  const float* gWxf2 = (const float*)d_in[56];
  const float* gWhf2 = (const float*)d_in[57];
  const float* gbd   = (const float*)d_in[58];
  const float* gbo   = (const float*)d_in[59];
  const float* gbf1  = (const float*)d_in[60];
  const float* gbf2  = (const float*)d_in[61];
  const float* gdng  = (const float*)d_in[62];
  const float* gdnb  = (const float*)d_in[63];
  const float* gong  = (const float*)d_in[64];
  const float* gonb  = (const float*)d_in[65];
  const float* gfng  = (const float*)d_in[66];
  const float* gfnb  = (const float*)d_in[67];

  // ---------------- scratch carve-out
  char* base = (char*)d_ws;
  size_t off = 0;
  auto carve = [&](size_t bytes) -> char* {
    off = (off + 255) & ~(size_t)255;
    char* p = base + off; off += bytes; return p;
  };
  unsigned short* WbigT  = (unsigned short*)carve((size_t)N6 * K4 * 2);
  unsigned short* WdbigT = (unsigned short*)carve((size_t)N6 * H * 2);
  unsigned short* WxT    = (unsigned short*)carve((size_t)4*H * H * 2);
  unsigned short* WhT    = (unsigned short*)carve((size_t)2*H * N6 * 2);
  unsigned short* Whf1T  = (unsigned short*)carve((size_t)H * H * 2);
  unsigned short* Whf2T  = (unsigned short*)carve((size_t)H * H * 2);
  unsigned short* X1     = (unsigned short*)carve((size_t)RSEQ * K4 * 2);
  unsigned short* X2     = (unsigned short*)carve((size_t)RSEQ * K4 * 2);
  unsigned short* ghcomb = (unsigned short*)carve((size_t)BB * (H + N6) * 2);
  float* Pbuf = (float*)carve((size_t)RSEQ * N6 * 4);
  float* h1b[2], *c1b[2], *h2b[2], *c2b[2], *ghb[2], *gcb[2];
  for (int i = 0; i < 2; ++i) {
    h1b[i] = (float*)carve((size_t)RSEQ * H * 4);
    c1b[i] = (float*)carve((size_t)RSEQ * H * 4);
    h2b[i] = (float*)carve((size_t)RSEQ * H * 4);
    c2b[i] = (float*)carve((size_t)RSEQ * H * 4);
    ghb[i] = (float*)carve((size_t)BB * H * 4);
    gcb[i] = (float*)carve((size_t)BB * H * 4);
  }
  float* emb1 = (float*)carve((size_t)RSEQ * H * 4);
  float* emb2 = (float*)carve((size_t)RSEQ * H * 4);
  float* gf1  = (float*)carve((size_t)RSEQ * H * 4);
  float* gf2  = (float*)carve((size_t)RSEQ * H * 4);
  float* Dbuf = (float*)carve((size_t)BB * N6 * 4);
  float* GLB  = (float*)carve((size_t)BB * 4*H * 4);
  float* gdv  = (float*)carve((size_t)BB * H * 4);
  float* gov  = (float*)carve((size_t)BB * H * 4);
  float* comb = (float*)carve((size_t)BB * N6 * 4);

  // ---------------- weight packing (bf16, [N,K] transposed)
  {
    P18 w;
    for (int g = 0; g < 6; ++g) { w.p[g*3+0] = Wx[g]; w.p[g*3+1] = Wh[g]; w.p[g*3+2] = Wi[g]; }
    int tot = N6 * K4;
    pack_wbigT<<<(tot + 255) / 256, 256, 0, stream>>>(WbigT, w);
  }
  {
    P6 w = {{ Wd[0], Wd[1], Wd[2], Wd[3], Wd[4], Wd[5] }};
    int tot = N6 * H;
    pack_colT<<<(tot + 255) / 256, 256, 0, stream>>>(WdbigT, w, N6, H);
  }
  {
    P6 w = {{ gWxd, gWxo, gWxf1, gWxf2, nullptr, nullptr }};
    int tot = 4*H * H;
    pack_colT<<<(tot + 255) / 256, 256, 0, stream>>>(WxT, w, 4*H, H);
  }
  {
    P6 w = {{ gWhd, gWho, nullptr, nullptr, nullptr, nullptr }};
    int tot = 2*H * N6;
    pack_colT<<<(tot + 255) / 256, 256, 0, stream>>>(WhT, w, 2*H, N6);
  }
  {
    P6 w = {{ gWhf1, nullptr, nullptr, nullptr, nullptr, nullptr }};
    pack_colT<<<(H*H + 255) / 256, 256, 0, stream>>>(Whf1T, w, H, H);
  }
  {
    P6 w = {{ gWhf2, nullptr, nullptr, nullptr, nullptr, nullptr }};
    pack_colT<<<(H*H + 255) / 256, 256, 0, stream>>>(Whf2T, w, H, H);
  }

  // ---------------- initial state
  int cur = 0;
  {
    int tot = RSEQ * H;
    init_state<<<(tot + 255) / 256, 256, 0, stream>>>(s1i, h10, c10, m1, h1b[cur], c1b[cur], emb1);
    init_state<<<(tot + 255) / 256, 256, 0, stream>>>(s2i, h20, c20, m2, h2b[cur], c2b[cur], emb2);
    init_global<<<(BB*H + 255) / 256, 256, 0, stream>>>(h1b[cur], h2b[cur], c1b[cur], c2b[cur],
                                                        ghb[cur], gcb[cur]);
  }

  // ---------------- layer loop
  for (int layer = 0; layer < 7; ++layer) {
    int nxt = 1 - cur;

    fusion_comb<<<(BB*H + 255) / 256, 256, 0, stream>>>(h1b[cur], m1, h2b[cur], m2, comb);
    {
      int tot = RSEQ * K4;
      build_X<<<(tot + 255) / 256, 256, 0, stream>>>(h1b[cur], emb1, X1);
      build_X<<<(tot + 255) / 256, 256, 0, stream>>>(h2b[cur], emb2, X2);
      int tg = BB * (H + N6);
      build_ghcomb<<<(tg + 255) / 256, 256, 0, stream>>>(ghb[cur], comb, ghcomb);
    }

    // global gate pre-acts: GLB[:, 0:4H] = g_h@[Wxd|Wxo|Wxf1|Wxf2]; GLB[:, 0:2H] += comb@[Whd|Who]
    gemm_bf16_wmma<<<gemm_grid(BB, 4*H), 256, 0, stream>>>(
        ghcomb, H + N6, WxT, GLB, 4*H, BB, 4*H, H, nullptr, 1, 1, 0);
    gemm_bf16_wmma<<<gemm_grid(BB, 2*H), 256, 0, stream>>>(
        ghcomb + H, H + N6, WhT, GLB, 4*H, BB, 2*H, N6, nullptr, 1, 1, 1);

    ln_sigmoid<<<BB, 256, 0, stream>>>(GLB, 4*H, 0,   gdv, H, 0, gdng, gdnb, gbd);
    ln_sigmoid<<<BB, 256, 0, stream>>>(GLB, 4*H, H,   gov, H, 0, gong, gonb, gbo);

    // gf1/gf2 = sigmoid(LN(h @ Whf + broadcast(g_h @ Wxf)))
    gemm_bf16_wmma<<<gemm_grid(RSEQ, H), 256, 0, stream>>>(
        X1, K4, Whf1T, gf1, H, RSEQ, H, H, GLB + 2*H, 4*H, SS, 0);
    ln_sigmoid<<<RSEQ, 256, 0, stream>>>(gf1, H, 0, gf1, H, 0, gfng, gfnb, gbf1);
    gemm_bf16_wmma<<<gemm_grid(RSEQ, H), 256, 0, stream>>>(
        X2, K4, Whf2T, gf2, H, RSEQ, H, H, GLB + 3*H, 4*H, SS, 0);
    ln_sigmoid<<<RSEQ, 256, 0, stream>>>(gf2, H, 0, gf2, H, 0, gfng, gfnb, gbf2);

    scores_update<<<(BB*H + 255) / 256, 256, 0, stream>>>(
        gf1, gf2, gdv, gov, m1, m2, c1b[cur], c2b[cur], gcb[cur], ghb[nxt], gcb[nxt]);

    // D = g_h @ [Wd_f1|...|Wd_o]
    gemm_bf16_wmma<<<gemm_grid(BB, N6), 256, 0, stream>>>(
        ghcomb, H + N6, WdbigT, Dbuf, N6, BB, N6, H, nullptr, 1, 1, 0);

    // sequence 1: fused gate GEMM + LN/sigmoid + cell update
    gemm_bf16_wmma<<<gemm_grid(RSEQ, N6), 256, 0, stream>>>(
        X1, K4, WbigT, Pbuf, N6, RSEQ, N6, K4, Dbuf, N6, SS, 0);
    for (int g = 0; g < 6; ++g)
      ln_sigmoid<<<RSEQ, 256, 0, stream>>>(Pbuf, N6, g*H, Pbuf, N6, g*H, ng[g], nb[g], bg[g]);
    slstm_update<<<(RSEQ*H + 255) / 256, 256, 0, stream>>>(
        Pbuf, emb1, c1b[cur], gcb[cur], m1, h1b[nxt], c1b[nxt]);

    // sequence 2
    gemm_bf16_wmma<<<gemm_grid(RSEQ, N6), 256, 0, stream>>>(
        X2, K4, WbigT, Pbuf, N6, RSEQ, N6, K4, Dbuf, N6, SS, 0);
    for (int g = 0; g < 6; ++g)
      ln_sigmoid<<<RSEQ, 256, 0, stream>>>(Pbuf, N6, g*H, Pbuf, N6, g*H, ng[g], nb[g], bg[g]);
    slstm_update<<<(RSEQ*H + 255) / 256, 256, 0, stream>>>(
        Pbuf, emb2, c2b[cur], gcb[cur], m2, h2b[nxt], c2b[nxt]);

    cur = nxt;
  }

  // ---------------- outputs: h1, h2, feats, g_h
  float* out = (float*)d_out;
  hipMemcpyAsync(out,                 h1b[cur], (size_t)RSEQ * H * 4, hipMemcpyDeviceToDevice, stream);
  hipMemcpyAsync(out + RSEQ*H,        h2b[cur], (size_t)RSEQ * H * 4, hipMemcpyDeviceToDevice, stream);
  fusion_comb<<<(BB*H + 255) / 256, 256, 0, stream>>>(h1b[cur], m1, h2b[cur], m2, out + 2*RSEQ*H);
  hipMemcpyAsync(out + 2*RSEQ*H + BB*N6, ghb[cur], (size_t)BB * H * 4, hipMemcpyDeviceToDevice, stream);
}
